// BoundedCompositionalEnergy_85950885528258
// MI455X (gfx1250) — matrix operand order repaired and checked by reading, not compiled, not run on hardware
//
#include <hip/hip_runtime.h>
#include <cstddef>

#define B_TOT     16384
#define IN_DIMS   1280
#define STATE_D   1024
#define ACTION_D  256
#define HDIM      1024
#define NC        8
#define KTILES    40          // 1280 / 32
#define MTILE     32          // rows per workgroup (2 x 16-row WMMA tiles)

typedef __attribute__((ext_vector_type(16))) __bf16 v16bf;
typedef __attribute__((ext_vector_type(8)))  __bf16 v8bf;
typedef __attribute__((ext_vector_type(8)))  float  v8f;

__device__ __forceinline__ unsigned short f2bf(float f) {
  unsigned u = __float_as_uint(f);
  u += 0x7FFFu + ((u >> 16) & 1u);          // round-to-nearest-even
  return (unsigned short)(u >> 16);
}
__device__ __forceinline__ float bf2f(unsigned short h) {
  return __uint_as_float(((unsigned)h) << 16);
}

// ---------------------------------------------------------------- pack x ----
__global__ void pack_x_kernel(const float* __restrict__ states,
                              const float* __restrict__ actions,
                              unsigned short* __restrict__ xb) {
  long long tid = (long long)blockIdx.x * blockDim.x + threadIdx.x;
  if (tid >= (long long)B_TOT * IN_DIMS) return;
  int i = (int)(tid % IN_DIMS);
  int b = (int)(tid / IN_DIMS);
  float v = (i < STATE_D) ? states[(size_t)b * STATE_D + i]
                          : actions[(size_t)b * ACTION_D + (i - STATE_D)];
  xb[tid] = f2bf(v);
}

// --------------------------------------------------------------- pack W1 ----
// w1t[c][kt][n][kk]: kk = 0..31 contiguous K within a K-tile, so a lane's
// B-fragment (16 contiguous bf16 at kk offset 0 or 16) is one 32B read.
__global__ void pack_w1_kernel(const float* __restrict__ W1,
                               unsigned short* __restrict__ w1t) {
  long long tid = (long long)blockIdx.x * blockDim.x + threadIdx.x;
  if (tid >= (long long)NC * KTILES * HDIM * 32) return;
  int n  = (int)(tid % HDIM);
  long long t2 = tid / HDIM;
  int kk = (int)(t2 % 32);  t2 /= 32;
  int kt = (int)(t2 % KTILES);
  int c  = (int)(t2 / KTILES);
  int k  = kt * 32 + kk;
  float v = W1[((size_t)c * IN_DIMS + k) * HDIM + n];
  w1t[(((size_t)((c * KTILES + kt) * HDIM + n)) << 5) + kk] = f2bf(v);
}

// ---------------------------------------------------------- weight logits ---
__global__ void logits_kernel(const unsigned short* __restrict__ xb,
                              const float* __restrict__ Ww,
                              const float* __restrict__ bw,
                              float* __restrict__ logits) {
  int tid = blockIdx.x * blockDim.x + threadIdx.x;   // B_TOT * NC threads
  int c = tid & (NC - 1);
  int b = tid >> 3;
  const unsigned short* xr = xb + (size_t)b * IN_DIMS;
  float acc = 0.f;
#pragma unroll 4
  for (int i = 0; i < IN_DIMS; ++i)
    acc += bf2f(xr[i]) * Ww[i * NC + c];
  logits[tid] = acc + bw[c];
}

// ------------------------------------------------ fused GEMM+LN+GELU+dot ----
// Grid (B/32, C); 256 threads = 8 wave32s, each wave owns a 128-col strip and
// TWO 16-row M tiles (one B fragment feeds two WMMAs -> half the L2 traffic).
// Dynamic LDS: phase 1 = 32x1280 bf16 x-tile (80 KB, filled by async-to-LDS),
// phase 2 = 16x1024 f32 h-tile (64 KB), aliased (x dead after the k-loop).
__global__ __launch_bounds__(256)
void moe_energy_main_kernel(const unsigned short* __restrict__ xb,
                            const unsigned short* __restrict__ w1t,
                            const float* __restrict__ b1,
                            const float* __restrict__ gamma,
                            const float* __restrict__ beta,
                            const float* __restrict__ W2,
                            const float* __restrict__ b2,
                            float* __restrict__ comps) {
  extern __shared__ unsigned char dynsmem[];
  unsigned short* sx = (unsigned short*)dynsmem;   // 32*1280 bf16 = 80 KB
  float*          sh = (float*)dynsmem;            // 16*1024 f32  = 64 KB

  const int tid  = threadIdx.x;
  const int wave = tid >> 5;
  const int lane = tid & 31;
  const int b0   = blockIdx.x * MTILE;
  const int c    = blockIdx.y;

  // ---- async-stage 32x1280 bf16 x-tile (contiguous 80 KB) into LDS --------
  {
    unsigned lbase = (unsigned)(size_t)sx;   // low 32 bits = LDS offset
    const char* g = (const char*)(xb + (size_t)b0 * IN_DIMS);
    for (int i = tid; i < (MTILE * IN_DIMS * 2) / 16; i += 256) {
      unsigned laddr = lbase + (unsigned)i * 16u;
      unsigned long long gaddr = (unsigned long long)(size_t)(g + (size_t)i * 16);
      asm volatile("global_load_async_to_lds_b128 %0, %1, off"
                   :: "v"(laddr), "v"(gaddr) : "memory");
    }
    asm volatile("s_wait_asynccnt 0x0" ::: "memory");
  }
  __syncthreads();

  v8f acc0[8] = {};   // rows 0..15  x 128 cols
  v8f acc1[8] = {};   // rows 16..31 x 128 cols
  const int halfSel = (lane < 16) ? 0 : 1;
  const int laneM   = lane & 15;
  const int nlane   = wave * 128 + laneM;

  const unsigned short* w1c = w1t + (((size_t)c * KTILES * HDIM) << 5);

  for (int kt = 0; kt < KTILES; ++kt) {
    // Two A fragments from LDS (contiguous K, matching B: lanes 0-15 hold
    // K{0..15}, lanes 16-31 hold K{16..31}).
    const unsigned short* xrow0 =
        sx + laneM * IN_DIMS + kt * 32 + (halfSel ? 16 : 0);
    union { v16bf v; v8bf h[2]; } a0, a1;
    a0.h[0] = *(const v8bf*)(xrow0);
    a0.h[1] = *(const v8bf*)(xrow0 + 8);
    a1.h[0] = *(const v8bf*)(xrow0 + 16 * IN_DIMS);
    a1.h[1] = *(const v8bf*)(xrow0 + 16 * IN_DIMS + 8);

    const unsigned short* wkt =
        w1c + (((size_t)kt * HDIM) << 5) + (halfSel ? 16 : 0);

    if (kt + 1 < KTILES)  // pull next K-tile of W1 toward the caches
      __builtin_prefetch(wkt + (((size_t)HDIM) << 5), 0, 0);

    // Software-pipelined B fragments: double-buffered registers so the next
    // global load is in flight while the current pair of WMMAs executes.
    v16bf bcur = *(const v16bf*)(wkt + ((size_t)nlane << 5));
#pragma unroll
    for (int t = 0; t < 8; ++t) {
      v16bf bnext = bcur;
      if (t < 7)
        bnext = *(const v16bf*)(wkt + ((size_t)(nlane + (t + 1) * 16) << 5));
      acc0[t] = __builtin_amdgcn_wmma_f32_16x16x32_bf16(
          false, a0.v, false, bcur, (short)0, acc0[t], false, false);
      acc1[t] = __builtin_amdgcn_wmma_f32_16x16x32_bf16(
          false, a1.v, false, bcur, (short)0, acc1[t], false, false);
      bcur = bnext;
    }
  }

  // ---- epilogue: two 16-row passes through the shared 64 KB h buffer ------
  float b1v[8];
#pragma unroll
  for (int t = 0; t < 8; ++t) b1v[t] = b1[c * HDIM + nlane + t * 16];

  const int row = tid >> 4;
  const int sub = tid & 15;

  auto epilogue = [&](v8f (&accp)[8], int prow) {
    __syncthreads();   // p0: x-tile reads done; p1: previous h reads done
#pragma unroll
    for (int t = 0; t < 8; ++t) {
      int n = nlane + t * 16;
#pragma unroll
      for (int r = 0; r < 8; ++r) {
        int m = r + halfSel * 8;                 // C/D VGPR layout
        sh[m * HDIM + n] = accp[t][r] + b1v[t];
      }
    }
    __syncthreads();

    // LayerNorm stats: 16 threads per row, butterfly reduce (wave32)
    float s = 0.f, sq = 0.f;
#pragma unroll 8
    for (int j = 0; j < 64; ++j) {
      float v = sh[row * HDIM + j * 16 + sub];
      s += v; sq += v * v;
    }
#pragma unroll
    for (int m = 8; m >= 1; m >>= 1) {
      s  += __shfl_xor(s,  m, 32);
      sq += __shfl_xor(sq, m, 32);
    }
    float mu   = s * (1.0f / HDIM);
    float var  = sq * (1.0f / HDIM) - mu * mu;
    float rstd = rsqrtf(var + 1e-5f);

    float dp = 0.f;
    for (int j = 0; j < 64; ++j) {
      int n = j * 16 + sub;
      float v  = sh[row * HDIM + n];
      float hn = (v - mu) * rstd * gamma[c * HDIM + n] + beta[c * HDIM + n];
      float g  = 0.5f * hn * (1.0f + erff(hn * 0.70710678118654752f));
      dp += g * W2[c * HDIM + n];
    }
#pragma unroll
    for (int m = 8; m >= 1; m >>= 1) dp += __shfl_xor(dp, m, 32);
    if (sub == 0) comps[(size_t)(b0 + prow + row) * NC + c] = dp + b2[c];
  };

  epilogue(acc0, 0);
  epilogue(acc1, 16);
}

// --------------------------------------------------- softmax-weighted sum ---
__global__ void energy_kernel(const float* __restrict__ logits,
                              const float* __restrict__ comps,
                              float* __restrict__ out) {
  int b = blockIdx.x * blockDim.x + threadIdx.x;
  if (b >= B_TOT) return;
  float l[NC], mx = -1e30f;
#pragma unroll
  for (int c = 0; c < NC; ++c) { l[c] = logits[b * NC + c]; mx = fmaxf(mx, l[c]); }
  float den = 0.f;
#pragma unroll
  for (int c = 0; c < NC; ++c) { l[c] = __expf(l[c] - mx); den += l[c]; }
  float e = 0.f;
#pragma unroll
  for (int c = 0; c < NC; ++c) e += l[c] * comps[b * NC + c];
  out[b] = e / den;
}

// ---------------------------------------------------------------- launch ----
extern "C" void kernel_launch(void* const* d_in, const int* in_sizes, int n_in,
                              void* d_out, int out_size, void* d_ws, size_t ws_size,
                              hipStream_t stream) {
  const float* states  = (const float*)d_in[0];
  const float* actions = (const float*)d_in[1];
  const float* W1      = (const float*)d_in[2];
  const float* b1      = (const float*)d_in[3];
  const float* gamma   = (const float*)d_in[4];
  const float* beta    = (const float*)d_in[5];
  const float* W2      = (const float*)d_in[6];
  const float* b2      = (const float*)d_in[7];
  const float* Ww      = (const float*)d_in[8];
  const float* bw      = (const float*)d_in[9];
  float* out = (float*)d_out;

  const size_t XB_BYTES  = (size_t)B_TOT * IN_DIMS * 2;            // 40 MB
  const size_t W1T_BYTES = (size_t)NC * KTILES * HDIM * 32 * 2;    // 20 MB
  unsigned short* xb  = (unsigned short*)d_ws;
  unsigned short* w1t = (unsigned short*)((char*)d_ws + XB_BYTES);
  float* logits = (float*)((char*)d_ws + XB_BYTES + W1T_BYTES);
  float* comps  = logits + (size_t)B_TOT * NC;

  pack_x_kernel<<<(B_TOT * IN_DIMS) / 256, 256, 0, stream>>>(states, actions, xb);
  pack_w1_kernel<<<(NC * KTILES * HDIM * 32) / 256, 256, 0, stream>>>(W1, w1t);
  logits_kernel<<<(B_TOT * NC) / 256, 256, 0, stream>>>(xb, Ww, bw, logits);

  dim3 grid(B_TOT / MTILE, NC);
  size_t lds_bytes = (size_t)MTILE * IN_DIMS * 2;                  // 80 KB
  moe_energy_main_kernel<<<grid, 256, lds_bytes, stream>>>(
      xb, w1t, b1, gamma, beta, W2, b2, comps);

  energy_kernel<<<B_TOT / 256, 256, 0, stream>>>(logits, comps, out);
}
